// MultiHeadAttention_48833778155734
// MI455X (gfx1250) — matrix operand order repaired and checked by reading, not compiled, not run on hardware
//
#include <hip/hip_runtime.h>

// ---------------------------------------------------------------------------
// MHA forward for MI455X (gfx1250, wave32, WMMA f32_16x16x32_f16 + TDM)
// B=4, T=2048, D_MODEL=1024, N_HEADS=16, D_HEAD=64
// ---------------------------------------------------------------------------

#define Bsz 4
#define Tsz 2048
#define Dm  1024
#define Hn  16
#define Dh  64

typedef __attribute__((ext_vector_type(16))) _Float16     v16h;
typedef __attribute__((ext_vector_type(8)))  _Float16     v8h;
typedef __attribute__((ext_vector_type(8)))  float        v8f;
typedef __attribute__((ext_vector_type(4)))  unsigned int v4u;
typedef __attribute__((ext_vector_type(8)))  int          v8i;
typedef __attribute__((ext_vector_type(4)))  int          v4i;

// ---------------- fragment loaders (layouts per CDNA5 ISA) -------

// A-matrix 16x32 f16: lane L holds row M = L&15.
// elements 0..7  -> K = koff + i   (koff = 0 lanes 0-15, 8 lanes 16-31)
// elements 8..15 -> K = koff + 16 + i
__device__ __forceinline__ v16h load_a_frag(const _Float16* base, int stride, int lane) {
    const int m    = lane & 15;
    const int koff = (lane & 16) ? 8 : 0;
    const _Float16* p = base + (size_t)m * stride + koff;
    v8h lo = *(const v8h*)(p);
    v8h hi = *(const v8h*)(p + 16);
    v16h r;
#pragma unroll
    for (int i = 0; i < 8; ++i) { r[i] = lo[i]; r[8 + i] = hi[i]; }
    return r;
}

// B-matrix 32x16 f16: lane L holds column N = L&15; elements 0..7 -> K = kgrp + i,
// elements 8..15 -> K = kgrp + 16 + i (kgrp = 0 lanes 0-15, 8 lanes 16-31). src row-major with N as row index.
__device__ __forceinline__ v16h load_b_frag(const _Float16* src, int stride, int lane) {
    const int n    = lane & 15;
    const int kgrp = (lane & 16) ? 8 : 0;
    const _Float16* p = src + (size_t)n * stride + kgrp;
    v8h lo = *(const v8h*)(p);
    v8h hi = *(const v8h*)(p + 16);
    v16h r;
#pragma unroll
    for (int i = 0; i < 8; ++i) { r[i] = lo[i]; r[8 + i] = hi[i]; }
    return r;
}

__device__ __forceinline__ v8f wmma_f16(v16h a, v16h b, v8f c) {
    return __builtin_amdgcn_wmma_f32_16x16x32_f16(false, a, false, b, (short)0, c,
                                                  false, false);
}

// ---------------- Tensor Data Mover: 2D f16 tile -> LDS ---------------------
// D# per CDNA5 ISA (count=1, type=2, data_size=2B,
// 2-D tile: tile1 rows x tile0 elems, row stride = stride0 elements).
__device__ __forceinline__ void tdm_load_2d_f16(const void* gptr, unsigned lds_off,
                                                unsigned dim0, unsigned dim1,
                                                unsigned stride0,
                                                unsigned tile0, unsigned tile1) {
    unsigned long long ga = (unsigned long long)gptr;
    v4u g0;
    g0[0] = 1u;                                   // count=1 (valid), user mode
    g0[1] = lds_off;                              // lds_addr (bytes)
    g0[2] = (unsigned)ga;                         // global_addr[31:0]
    g0[3] = (unsigned)(ga >> 32) | (2u << 30);    // global_addr[56:32] | type=2
    v8i g1;
    g1[0] = 0x00010000;                           // workgroup_mask=0, data_size=2B
    g1[1] = (int)((dim0 & 0xFFFFu) << 16);        // tensor_dim0[15:0]
    g1[2] = (int)((dim0 >> 16) | ((dim1 & 0xFFFFu) << 16));  // dim0 hi | dim1 lo
    g1[3] = (int)((dim1 >> 16) | (tile0 << 16));  // dim1 hi | tile_dim0
    g1[4] = (int)tile1;                           // tile_dim1 | tile_dim2=0
    g1[5] = (int)stride0;                         // tensor_dim0_stride[31:0]
    g1[6] = 0;                                    // stride0 hi | stride1 lo
    g1[7] = 0;                                    // stride1 hi
    v4i z4 = {0, 0, 0, 0};
#if __has_include(<hip/amd_detail/amd_gfx1250_TDM.h>)
    v8i z8 = {0, 0, 0, 0, 0, 0, 0, 0};
    __builtin_amdgcn_tensor_load_to_lds(g0, g1, z4, z4, z8, 0);
#else
    __builtin_amdgcn_tensor_load_to_lds(g0, g1, z4, z4, 0);
#endif
}

__device__ __forceinline__ unsigned lds_offset_of(const void* p) {
    return (unsigned)(unsigned long long)p;       // LDS aperture: addr[31:0] = offset
}

// 16-lane-group reductions (wave32: masks < 16 stay inside each half)
__device__ __forceinline__ float grp_max(float v) {
    v = fmaxf(v, __shfl_xor(v, 1, 32));
    v = fmaxf(v, __shfl_xor(v, 2, 32));
    v = fmaxf(v, __shfl_xor(v, 4, 32));
    v = fmaxf(v, __shfl_xor(v, 8, 32));
    return v;
}
__device__ __forceinline__ float grp_sum(float v) {
    v += __shfl_xor(v, 1, 32);
    v += __shfl_xor(v, 2, 32);
    v += __shfl_xor(v, 4, 32);
    v += __shfl_xor(v, 8, 32);
    return v;
}

// ---------------- prep: fp32 -> f16 conversions -----------------------------

__global__ void cvt_f32_to_f16(const float* __restrict__ src,
                               _Float16* __restrict__ dst, int n) {
    int i = blockIdx.x * blockDim.x + threadIdx.x;
    if (i < n) dst[i] = (_Float16)src[i];
}

// dst[n*K + k] = src[k*N + n]
__global__ void cvt_transpose_f16(const float* __restrict__ src,
                                  _Float16* __restrict__ dst, int N, int K) {
    int i = blockIdx.x * blockDim.x + threadIdx.x;
    if (i >= N * K) return;
    int n = i / K, k = i - n * K;
    dst[i] = (_Float16)src[(size_t)k * N + n];
}

// ---------------- QKV GEMM: (8192 x 1024) x (1024 x 3072) -------------------
// Block = 8 waves, tile 128(M) x 64(N). TDM double-buffers the A (128x32) and
// B (64x32) K-panels into LDS; waves consume via ds_load fragments + WMMA.

__global__ void __launch_bounds__(256)
qkv_gemm(const _Float16* __restrict__ X,
         const _Float16* __restrict__ Wt,  // [3072][1024]
         _Float16* __restrict__ Q,         // [B][H][T][Dh]
         _Float16* __restrict__ Km,        // [B][H][T][Dh]
         _Float16* __restrict__ Vt) {      // [B][H][Dh][T]
    __shared__ __align__(16) _Float16 As[2][128 * 32];
    __shared__ __align__(16) _Float16 Bs[2][64 * 32];

    const int lane = threadIdx.x & 31;
    const int w    = threadIdx.x >> 5;
    const int nblk = blockIdx.x % 48;
    const int mblk = blockIdx.x / 48;
    const int m0   = mblk * 128;
    const int nb   = nblk * 64;

    v8f acc[4];
#pragma unroll
    for (int t = 0; t < 4; ++t)
#pragma unroll
        for (int j = 0; j < 8; ++j) acc[t][j] = 0.0f;

    const _Float16* Ab = X  + (size_t)m0 * Dm;
    const _Float16* Bb = Wt + (size_t)nb * Dm;

    if (threadIdx.x < 32) {   // wave 0 drives the TDM
        tdm_load_2d_f16(Ab, lds_offset_of(&As[0][0]), Dm, 8192, Dm, 32, 128);
        tdm_load_2d_f16(Bb, lds_offset_of(&Bs[0][0]), Dm, 3072, Dm, 32, 64);
    }

    const int NSTEP = Dm / 32;
    for (int i = 0; i < NSTEP; ++i) {
        const int cur = i & 1;
        __syncthreads();                          // everyone done with buf cur^1
        if (threadIdx.x < 32) {
            if (i + 1 < NSTEP) {
                tdm_load_2d_f16(Ab + (i + 1) * 32, lds_offset_of(&As[cur ^ 1][0]),
                                Dm, 8192, Dm, 32, 128);
                tdm_load_2d_f16(Bb + (i + 1) * 32, lds_offset_of(&Bs[cur ^ 1][0]),
                                Dm, 3072, Dm, 32, 64);
                __builtin_amdgcn_s_wait_tensorcnt(2);   // buf cur complete
            } else {
                __builtin_amdgcn_s_wait_tensorcnt(0);
            }
        }
        __syncthreads();                          // release waves on buf cur

        v16h a = load_a_frag(&As[cur][w * 16 * 32], 32, lane);
#pragma unroll
        for (int t = 0; t < 4; ++t) {
            v16h b = load_b_frag(&Bs[cur][t * 16 * 32], 32, lane);
            acc[t] = wmma_f16(a, b, acc[t]);
        }
    }

    const int hi = (lane >> 4) & 1;
#pragma unroll
    for (int t = 0; t < 4; ++t) {
        const int ncol = nb + t * 16 + (lane & 15);
#pragma unroll
        for (int j = 0; j < 8; ++j) {
            const int M  = m0 + w * 16 + j + hi * 8;   // global row = b*T + t
            const int bb = M >> 11;
            const int tt = M & (Tsz - 1);
            _Float16 hv = (_Float16)acc[t][j];
            if (ncol < Dm) {
                int h = ncol >> 6, d = ncol & 63;
                Q[(((size_t)(bb * Hn + h)) * Tsz + tt) * Dh + d] = hv;
            } else if (ncol < 2 * Dm) {
                int f = ncol - Dm, h = f >> 6, d = f & 63;
                Km[(((size_t)(bb * Hn + h)) * Tsz + tt) * Dh + d] = hv;
            } else {
                int f = ncol - 2 * Dm, h = f >> 6, d = f & 63;
                Vt[(((size_t)(bb * Hn + h)) * Dh + d) * Tsz + tt] = hv;
            }
        }
    }
}

// ---------------- causal flash attention ------------------------------------
// wave = one (b, h, 16-query tile); online softmax over 32-key blocks.

__global__ void __launch_bounds__(256)
attn_fwd(const _Float16* __restrict__ Q,
         const _Float16* __restrict__ Km,
         const _Float16* __restrict__ Vt,
         _Float16* __restrict__ Y) {       // [B][T][H][Dh]
    __shared__ __align__(16) _Float16 ptile[8][16 * 32];

    const int lane = threadIdx.x & 31;
    const int w    = threadIdx.x >> 5;
    const int idx  = blockIdx.x * 8 + w;       // 0 .. 8191
    const int qt   = idx & 127;
    const int h    = (idx >> 7) & 15;
    const int b    = idx >> 11;
    const int q0   = qt * 16;
    const int hi   = (lane >> 4) & 1;

    const _Float16* Qp  = Q  + (size_t)(b * Hn + h) * Tsz * Dh;
    const _Float16* Kp  = Km + (size_t)(b * Hn + h) * Tsz * Dh;
    const _Float16* Vtp = Vt + (size_t)(b * Hn + h) * Dh * Tsz;

    v16h qf0 = load_a_frag(Qp + (size_t)q0 * Dh + 0,  Dh, lane);
    v16h qf1 = load_a_frag(Qp + (size_t)q0 * Dh + 32, Dh, lane);

    v8f o[4];
    float mrow[8], lrow[8];
#pragma unroll
    for (int t = 0; t < 4; ++t)
#pragma unroll
        for (int j = 0; j < 8; ++j) o[t][j] = 0.0f;
#pragma unroll
    for (int j = 0; j < 8; ++j) { mrow[j] = -1e30f; lrow[j] = 0.0f; }

    _Float16* pw = &ptile[w][0];

    for (int kb = 0; kb < q0 + 16; kb += 32) {
        __builtin_prefetch(Kp + (size_t)(kb + 32) * Dh, 0, 0);
        // ---- scores: two 16x16 S tiles, each contracted over d=64 ----------
        v8f s0, s1;
#pragma unroll
        for (int j = 0; j < 8; ++j) { s0[j] = 0.0f; s1[j] = 0.0f; }
        {
            v16h bk = load_b_frag(Kp + (size_t)(kb +  0) * Dh + 0,  Dh, lane);
            s0 = wmma_f16(qf0, bk, s0);
            bk = load_b_frag(Kp + (size_t)(kb +  0) * Dh + 32, Dh, lane);
            s0 = wmma_f16(qf1, bk, s0);
            bk = load_b_frag(Kp + (size_t)(kb + 16) * Dh + 0,  Dh, lane);
            s1 = wmma_f16(qf0, bk, s1);
            bk = load_b_frag(Kp + (size_t)(kb + 16) * Dh + 32, Dh, lane);
            s1 = wmma_f16(qf1, bk, s1);
        }

        // ---- scale + causal mask + online softmax --------------------------
        const int key0 = kb + (lane & 15);
        const int key1 = key0 + 16;
#pragma unroll
        for (int j = 0; j < 8; ++j) {
            const int qrow = q0 + j + hi * 8;
            float v0 = s0[j] * 0.125f;             // 1/sqrt(64)
            float v1 = s1[j] * 0.125f;
            if (key0 > qrow) v0 = -1e30f;
            if (key1 > qrow) v1 = -1e30f;

            float rm   = grp_max(fmaxf(v0, v1));
            float mnew = fmaxf(mrow[j], rm);
            float fac  = __expf(mrow[j] - mnew);
            float p0   = __expf(v0 - mnew);
            float p1   = __expf(v1 - mnew);
            float rs   = grp_sum(p0 + p1);
            lrow[j]    = lrow[j] * fac + rs;
            mrow[j]    = mnew;
#pragma unroll
            for (int t = 0; t < 4; ++t) o[t][j] *= fac;

            const int r = j + hi * 8;              // S/D-tile row
            pw[r * 32 + (lane & 15)]      = (_Float16)p0;
            pw[r * 32 + 16 + (lane & 15)] = (_Float16)p1;
        }

        // ---- restage P (16x32) into A-fragment layout via LDS --------------
        v16h pa = load_a_frag(pw, 32, lane);

        // ---- O += P x V (V transposed: B-frag rows contiguous) -------------
#pragma unroll
        for (int t = 0; t < 4; ++t) {
            v16h bv = load_b_frag(Vtp + (size_t)(t * 16) * Tsz + kb, Tsz, lane);
            o[t] = wmma_f16(pa, bv, o[t]);
        }
    }

    // ---- normalize + store Y[b][t][h*64+d] as f16 --------------------------
#pragma unroll
    for (int j = 0; j < 8; ++j) {
        const float inv = 1.0f / lrow[j];
        const int tq = q0 + j + hi * 8;
        _Float16* yrow = Y + ((size_t)(b * Tsz + tq)) * Dm + h * Dh;
#pragma unroll
        for (int t = 0; t < 4; ++t)
            yrow[t * 16 + (lane & 15)] = (_Float16)(o[t][j] * inv);
    }
}

// ---------------- output projection: (8192 x 1024) x (1024 x 1024) ----------

__global__ void __launch_bounds__(256)
out_gemm(const _Float16* __restrict__ Yf,
         const _Float16* __restrict__ Wt2,  // [1024][1024]
         float* __restrict__ Out) {
    __shared__ __align__(16) _Float16 As[2][128 * 32];
    __shared__ __align__(16) _Float16 Bs[2][64 * 32];

    const int lane = threadIdx.x & 31;
    const int w    = threadIdx.x >> 5;
    const int nblk = blockIdx.x % 16;
    const int mblk = blockIdx.x / 16;
    const int m0   = mblk * 128;
    const int nb   = nblk * 64;

    v8f acc[4];
#pragma unroll
    for (int t = 0; t < 4; ++t)
#pragma unroll
        for (int j = 0; j < 8; ++j) acc[t][j] = 0.0f;

    const _Float16* Ab = Yf  + (size_t)m0 * Dm;
    const _Float16* Bb = Wt2 + (size_t)nb * Dm;

    if (threadIdx.x < 32) {
        tdm_load_2d_f16(Ab, lds_offset_of(&As[0][0]), Dm, 8192, Dm, 32, 128);
        tdm_load_2d_f16(Bb, lds_offset_of(&Bs[0][0]), Dm, 1024, Dm, 32, 64);
    }

    const int NSTEP = Dm / 32;
    for (int i = 0; i < NSTEP; ++i) {
        const int cur = i & 1;
        __syncthreads();
        if (threadIdx.x < 32) {
            if (i + 1 < NSTEP) {
                tdm_load_2d_f16(Ab + (i + 1) * 32, lds_offset_of(&As[cur ^ 1][0]),
                                Dm, 8192, Dm, 32, 128);
                tdm_load_2d_f16(Bb + (i + 1) * 32, lds_offset_of(&Bs[cur ^ 1][0]),
                                Dm, 1024, Dm, 32, 64);
                __builtin_amdgcn_s_wait_tensorcnt(2);
            } else {
                __builtin_amdgcn_s_wait_tensorcnt(0);
            }
        }
        __syncthreads();

        v16h a = load_a_frag(&As[cur][w * 16 * 32], 32, lane);
#pragma unroll
        for (int t = 0; t < 4; ++t) {
            v16h b = load_b_frag(&Bs[cur][t * 16 * 32], 32, lane);
            acc[t] = wmma_f16(a, b, acc[t]);
        }
    }

    const int hi = (lane >> 4) & 1;
#pragma unroll
    for (int t = 0; t < 4; ++t) {
        const int ncol = nb + t * 16 + (lane & 15);
#pragma unroll
        for (int j = 0; j < 8; ++j) {
            const int M = m0 + w * 16 + j + hi * 8;
            Out[(size_t)M * Dm + ncol] = acc[t][j];
        }
    }
}

// ---------------- launch ----------------------------------------------------

extern "C" void kernel_launch(void* const* d_in, const int* in_sizes, int n_in,
                              void* d_out, int out_size, void* d_ws, size_t ws_size,
                              hipStream_t stream) {
    const float* x     = (const float*)d_in[0];   // [4][2048][1024]
    const float* W_qkv = (const float*)d_in[1];   // [1024][3072]
    const float* W_out = (const float*)d_in[2];   // [1024][1024]
    float* out = (float*)d_out;                   // [4][2048][1024]

    char* ws = (char*)d_ws;
    _Float16* Xf  = (_Float16*)(ws);                          // 16 MB
    _Float16* Wqt = (_Float16*)(ws + (size_t)16  * (1u<<20)); //  6 MB
    _Float16* Wot = (_Float16*)(ws + (size_t)22  * (1u<<20)); //  2 MB
    _Float16* Qb  = (_Float16*)(ws + (size_t)24  * (1u<<20)); // 16 MB
    _Float16* Kb  = (_Float16*)(ws + (size_t)40  * (1u<<20)); // 16 MB
    _Float16* Vtb = (_Float16*)(ws + (size_t)56  * (1u<<20)); // 16 MB
    _Float16* Yb  = (_Float16*)(ws + (size_t)72  * (1u<<20)); // 16 MB

    const int nX = Bsz * Tsz * Dm;                // 8388608
    cvt_f32_to_f16<<<(nX + 255) / 256, 256, 0, stream>>>(x, Xf, nX);
    cvt_transpose_f16<<<(3 * Dm * Dm + 255) / 256, 256, 0, stream>>>(W_qkv, Wqt, 3 * Dm, Dm);
    cvt_transpose_f16<<<(Dm * Dm + 255) / 256, 256, 0, stream>>>(W_out, Wot, Dm, Dm);

    // 64 M-blocks x 48 N-blocks, 128x64 per block
    qkv_gemm<<<3072, 256, 0, stream>>>(Xf, Wqt, Qb, Kb, Vtb);

    // 4*16*128 = 8192 waves / 8 per block
    attn_fwd<<<1024, 256, 0, stream>>>(Qb, Kb, Vtb, Yb);

    // 64 M-blocks x 16 N-blocks
    out_gemm<<<1024, 256, 0, stream>>>(Yb, Wot, out);
}
